// CausalSelfAttention_20177756356990
// MI455X (gfx1250) — matrix operand order repaired and checked by reading, not compiled, not run on hardware
//
#include <hip/hip_runtime.h>
#include <hip/hip_bf16.h>

typedef __attribute__((ext_vector_type(16))) _Float16 v16h;
typedef __attribute__((ext_vector_type(8)))  _Float16 v8h;
typedef __attribute__((ext_vector_type(8)))  float    v8f;

#define B_  4
#define T_  2048
#define C_  1024
#define H_  16
#define D_  64
#define M_  (B_ * T_)        // 8192 rows

// ---------------------------------------------------------------------------
// WMMA helpers (gfx1250, wave32). D(16x16,f32) = A(16x32,f16) x B(32x16,f16) + C
// A frag: lane (l&15) = row M, halves cover K = {h*8..+7, h*8+16..+23}, h=l>>4
// B frag: lane (l&15) = col N, halves cover K = {h*8..+7, h*8+16..+23}, h=l>>4
// C/D frag: elem r -> row r + 8*(l>>4), col = l&15
// ---------------------------------------------------------------------------
static __device__ inline v8f wmma_f16(v16h a, v16h b, v8f c) {
  return __builtin_amdgcn_wmma_f32_16x16x32_f16(false, a, false, b, (short)0, c, false, false);
}

static __device__ inline v16h load_a_frag(const _Float16* __restrict__ tile, int lda) {
  const int lane = threadIdx.x & 31, lm = lane & 15, hi = lane >> 4;
  const _Float16* p = tile + lm * lda + hi * 8;
  v8h lo = *(const v8h*)(p);
  v8h hh = *(const v8h*)(p + 16);
  v16h a;
#pragma unroll
  for (int i = 0; i < 8; ++i) { a[i] = lo[i]; a[i + 8] = hh[i]; }
  return a;
}

static __device__ inline v16h load_b_frag(const _Float16* __restrict__ tile, int ldb) {
  const int lane = threadIdx.x & 31, lm = lane & 15, hi = lane >> 4;
  const _Float16* p = tile + lm * ldb + hi * 8;
  v8h lo = *(const v8h*)(p);
  v8h hh = *(const v8h*)(p + 16);
  v16h b;
#pragma unroll
  for (int i = 0; i < 8; ++i) { b[i] = lo[i]; b[i + 8] = hh[i]; }
  return b;
}

// ---------------------------------------------------------------------------
// Kernel 1: fp32 -> fp16 streaming convert (n divisible by 4)
// ---------------------------------------------------------------------------
__global__ void cvt_f32_f16_k(const float* __restrict__ s, _Float16* __restrict__ d, int n) {
  int i = (blockIdx.x * blockDim.x + threadIdx.x) * 4;
  if (i < n) {
    float4 v = *(const float4*)(s + i);
    d[i + 0] = (_Float16)v.x;
    d[i + 1] = (_Float16)v.y;
    d[i + 2] = (_Float16)v.z;
    d[i + 3] = (_Float16)v.w;
  }
}

// ---------------------------------------------------------------------------
// GEMM core: 16x64 strip over K = C_. Unrolled-by-2 steady state with two
// NAMED buffer sets (a0/b0, a1/b1) that alternate -> zero rotation copies,
// and a dynamic loop so the scheduler keeps loads a full stage ahead of the
// consuming WMMAs (staggered s_wait_loadcnt instead of full drains).
// ---------------------------------------------------------------------------
static __device__ inline void gemm_16x64(const _Float16* __restrict__ arow,
                                         const _Float16* __restrict__ brow,
                                         v8f acc[4]) {
  v16h a0 = load_a_frag(arow, C_);
  v16h b0[4];
#pragma unroll
  for (int j = 0; j < 4; ++j) b0[j] = load_b_frag(brow + (j * 16) * C_, C_);

#pragma unroll 1
  for (int k = 0; k < C_; k += 64) {
    // stage A: fetch k+32 into buffer set 1, then consume buffer set 0
    v16h a1 = load_a_frag(arow + k + 32, C_);
    v16h b1[4];
#pragma unroll
    for (int j = 0; j < 4; ++j) b1[j] = load_b_frag(brow + (j * 16) * C_ + k + 32, C_);
    __builtin_prefetch(arow + k + 256, 0, 3);   // near-cache prefetch, 4 steps ahead
    __builtin_prefetch(brow + k + 256, 0, 3);
#pragma unroll
    for (int j = 0; j < 4; ++j) acc[j] = wmma_f16(a0, b0[j], acc[j]);

    // stage B: fetch k+64 into buffer set 0 (clamped on final trip), consume set 1
    const int kn = (k + 64 < C_) ? (k + 64) : 0;
    a0 = load_a_frag(arow + kn, C_);
#pragma unroll
    for (int j = 0; j < 4; ++j) b0[j] = load_b_frag(brow + (j * 16) * C_ + kn, C_);
#pragma unroll
    for (int j = 0; j < 4; ++j) acc[j] = wmma_f16(a1, b1[j], acc[j]);
  }
}

// ---------------------------------------------------------------------------
// Kernel 2: fused QKV projection. grid.y = matrix id (0=q,1=k,2=v).
// One wave -> 16 rows x 64 cols. Q,K stored [B,H,T,D]; V transposed [B,H,D,T].
// ---------------------------------------------------------------------------
__global__ __launch_bounds__(128) void qkv_kernel(
    const _Float16* __restrict__ x16,
    const _Float16* __restrict__ wq, const _Float16* __restrict__ wk, const _Float16* __restrict__ wv,
    const float* __restrict__ bq, const float* __restrict__ bk, const float* __restrict__ bv,
    const float* __restrict__ rope,      // [T][D/2][2]
    _Float16* __restrict__ Q, _Float16* __restrict__ Kc, _Float16* __restrict__ Vt) {
  const int mat  = blockIdx.y;
  const _Float16* w   = (mat == 0) ? wq : (mat == 1) ? wk : wv;
  const float*    bia = (mat == 0) ? bq : (mat == 1) ? bk : bv;

  const int wgid  = blockIdx.x * (blockDim.x >> 5) + (threadIdx.x >> 5);
  const int NB64  = C_ / 64;
  const int m0    = (wgid / NB64) * 16;
  const int n0    = (wgid % NB64) * 64;

  v8f acc[4] = {};
  gemm_16x64(x16 + m0 * C_, w + n0 * C_, acc);

  const int lane = threadIdx.x & 31, lm = lane & 15, hi = lane >> 4;
#pragma unroll
  for (int j = 0; j < 4; ++j) {
#pragma unroll
    for (int r = 0; r < 8; ++r) {
      const int row = m0 + r + 8 * hi;         // global row in [0, B*T)
      const int col = n0 + j * 16 + lm;        // global col in [0, C)
      float val = acc[j][r] + bia[col];
      const int t = row & (T_ - 1);
      const int d = col & (D_ - 1);
      if (mat < 2) {                           // RoPE on q,k
        const float* rp = rope + (t * (D_ / 2) + (d >> 1)) * 2;
        const float c = rp[0], sn = rp[1];
        const float other = __shfl_xor(val, 1, 32);   // neighbor column
        val = ((d & 1) == 0) ? (val * c - other * sn)
                             : (other * sn + val * c);
        if (mat == 0) val *= 0.125f;                  // fold 1/sqrt(D) into q
      }
      const int b  = row >> 11;
      const int h  = col >> 6;
      const int bh = b * H_ + h;
      if (mat == 0)      Q [(bh * T_ + t) * D_ + d] = (_Float16)val;
      else if (mat == 1) Kc[(bh * T_ + t) * D_ + d] = (_Float16)val;
      else               Vt[(bh * D_ + d) * T_ + t] = (_Float16)val;
    }
  }
}

// ---------------------------------------------------------------------------
// Kernel 3: causal flash attention. One wave per 16-row q tile; 32 keys/step.
// K-fragments double-buffered across chunks (unroll-2 renames most rotations);
// V-fragments issued early so softmax VALU hides their latency. P bounced
// through LDS (C/D-layout -> A-layout) with explicit s_wait_dscnt.
// ---------------------------------------------------------------------------
__global__ __launch_bounds__(128) void attn_kernel(
    const _Float16* __restrict__ Q, const _Float16* __restrict__ Kc,
    const _Float16* __restrict__ Vt, _Float16* __restrict__ Y) {
  __shared__ _Float16 lds[4 * 16 * 32];      // 1 KB per wave
  const int widx = threadIdx.x >> 5;
  const int lane = threadIdx.x & 31, lm = lane & 15, hi = lane >> 4;
  const int wg   = blockIdx.x * 4 + widx;
  const int QT   = T_ / 16;
  const int bh   = wg / QT;
  const int q0   = (wg % QT) * 16;

  const _Float16* Qb = Q  + (size_t)bh * T_ * D_;
  const _Float16* Kb = Kc + (size_t)bh * T_ * D_;
  const _Float16* Vb = Vt + (size_t)bh * D_ * T_;
  _Float16* pl = lds + widx * 512;

  const v16h aq0 = load_a_frag(Qb + q0 * D_ +  0, D_);
  const v16h aq1 = load_a_frag(Qb + q0 * D_ + 32, D_);

  v8f o[4] = {};
  float mr[8], lr[8];
#pragma unroll
  for (int r = 0; r < 8; ++r) { mr[r] = -1e30f; lr[r] = 0.f; }

  auto load_bk = [&](int kk, v16h bkf[4]) {
    bkf[0] = load_b_frag(Kb + (size_t)(kk     ) * D_ +  0, D_);
    bkf[1] = load_b_frag(Kb + (size_t)(kk     ) * D_ + 32, D_);
    bkf[2] = load_b_frag(Kb + (size_t)(kk + 16) * D_ +  0, D_);
    bkf[3] = load_b_frag(Kb + (size_t)(kk + 16) * D_ + 32, D_);
  };

  const int kend = q0 + 16;
  v16h bkc[4];
  load_bk(0, bkc);

#pragma unroll 2
  for (int k0 = 0; k0 < kend; k0 += 32) {
    // V fragments for this chunk: issue now, consume after softmax
    v16h bv[4];
#pragma unroll
    for (int j = 0; j < 4; ++j)
      bv[j] = load_b_frag(Vb + (size_t)(j * 16) * T_ + k0, T_);

    // S tiles from pre-fetched K fragments
    v8f s0 = {}, s1 = {};
    s0 = wmma_f16(aq0, bkc[0], s0);
    s0 = wmma_f16(aq1, bkc[1], s0);
    s1 = wmma_f16(aq0, bkc[2], s1);
    s1 = wmma_f16(aq1, bkc[3], s1);

    // prefetch next chunk's K fragments (clamped; redundant on last iter)
    const int k0n = (k0 + 32 < kend) ? (k0 + 32) : k0;
    v16h bkn[4];
    load_bk(k0n, bkn);

    // causal mask — only trailing chunks cross the diagonal
    if (k0 + 32 > q0) {
#pragma unroll
      for (int r = 0; r < 8; ++r) {
        const int row = q0 + r + 8 * hi;
        if (k0 + lm      > row) s0[r] = -1e30f;
        if (k0 + 16 + lm > row) s1[r] = -1e30f;
      }
    }

    // online softmax: per-row max/sum across the 16 lanes of each half
    float p0[8], p1[8];
#pragma unroll
    for (int r = 0; r < 8; ++r) {
      float cm = fmaxf(s0[r], s1[r]);
      cm = fmaxf(cm, __shfl_xor(cm, 1, 32));
      cm = fmaxf(cm, __shfl_xor(cm, 2, 32));
      cm = fmaxf(cm, __shfl_xor(cm, 4, 32));
      cm = fmaxf(cm, __shfl_xor(cm, 8, 32));
      const float mn = fmaxf(mr[r], cm);
      const float sc = __expf(mr[r] - mn);
      p0[r] = __expf(s0[r] - mn);
      p1[r] = __expf(s1[r] - mn);
      float rs = p0[r] + p1[r];
      rs += __shfl_xor(rs, 1, 32);
      rs += __shfl_xor(rs, 2, 32);
      rs += __shfl_xor(rs, 4, 32);
      rs += __shfl_xor(rs, 8, 32);
      lr[r] = lr[r] * sc + rs;
      mr[r] = mn;
#pragma unroll
      for (int j = 0; j < 4; ++j) o[j][r] *= sc;
      pl[(r + 8 * hi) * 32 + lm     ] = (_Float16)p0[r];
      pl[(r + 8 * hi) * 32 + 16 + lm] = (_Float16)p1[r];
    }
    asm volatile("s_wait_dscnt 0" ::: "memory");

    // O += P V
    const v16h ap = load_a_frag(pl, 32);
#pragma unroll
    for (int j = 0; j < 4; ++j) o[j] = wmma_f16(ap, bv[j], o[j]);

#pragma unroll
    for (int j = 0; j < 4; ++j) bkc[j] = bkn[j];
  }

  const int b = bh >> 4, h = bh & 15;
#pragma unroll
  for (int j = 0; j < 4; ++j) {
#pragma unroll
    for (int r = 0; r < 8; ++r) {
      const int t = q0 + r + 8 * hi;
      const int d = j * 16 + lm;
      Y[((size_t)(b * T_ + t)) * C_ + h * D_ + d] = (_Float16)(o[j][r] / lr[r]);
    }
  }
}

// ---------------------------------------------------------------------------
// Kernel 4: output projection out = Y @ Wp^T + bp (fp32 result)
// ---------------------------------------------------------------------------
__global__ __launch_bounds__(128) void proj_kernel(
    const _Float16* __restrict__ Y, const _Float16* __restrict__ wp,
    const float* __restrict__ bp, float* __restrict__ out) {
  const int wgid  = blockIdx.x * (blockDim.x >> 5) + (threadIdx.x >> 5);
  const int NB64  = C_ / 64;
  const int m0    = (wgid / NB64) * 16;
  const int n0    = (wgid % NB64) * 64;

  v8f acc[4] = {};
  gemm_16x64(Y + m0 * C_, wp + n0 * C_, acc);

  const int lane = threadIdx.x & 31, lm = lane & 15, hi = lane >> 4;
#pragma unroll
  for (int j = 0; j < 4; ++j)
#pragma unroll
    for (int r = 0; r < 8; ++r) {
      const int row = m0 + r + 8 * hi;
      const int col = n0 + j * 16 + lm;
      out[(size_t)row * C_ + col] = acc[j][r] + bp[col];
    }
}

// ---------------------------------------------------------------------------
extern "C" void kernel_launch(void* const* d_in, const int* in_sizes, int n_in,
                              void* d_out, int out_size, void* d_ws, size_t ws_size,
                              hipStream_t stream) {
  const float* x    = (const float*)d_in[0];
  const float* Wq   = (const float*)d_in[1];
  const float* bq   = (const float*)d_in[2];
  const float* Wk   = (const float*)d_in[3];
  const float* bk   = (const float*)d_in[4];
  const float* Wv   = (const float*)d_in[5];
  const float* bv   = (const float*)d_in[6];
  const float* Wp   = (const float*)d_in[7];
  const float* bp   = (const float*)d_in[8];
  const float* rope = (const float*)d_in[9];
  float* out = (float*)d_out;

  _Float16* W = (_Float16*)d_ws;
  const size_t nX = (size_t)M_ * C_;        // 8,388,608
  const size_t nW = (size_t)C_ * C_;        // 1,048,576
  _Float16* x16  = W;
  _Float16* wq16 = x16 + nX;
  _Float16* wk16 = wq16 + nW;
  _Float16* wv16 = wk16 + nW;
  _Float16* wp16 = wv16 + nW;
  _Float16* Qh   = wp16 + nW;               // [B,H,T,D]
  _Float16* Kh   = Qh + nX;                 // [B,H,T,D]
  _Float16* Vt   = Kh + nX;                 // [B,H,D,T]
  _Float16* Yh   = Vt + nX;                 // [B,T,C]
  const size_t need = (size_t)(Yh + nX - W) * sizeof(_Float16);   // ~92 MB
  if (ws_size < need) return;

  cvt_f32_f16_k<<<(int)(nX / 1024), 256, 0, stream>>>(x,  x16,  (int)nX);
  cvt_f32_f16_k<<<(int)(nW / 1024), 256, 0, stream>>>(Wq, wq16, (int)nW);
  cvt_f32_f16_k<<<(int)(nW / 1024), 256, 0, stream>>>(Wk, wk16, (int)nW);
  cvt_f32_f16_k<<<(int)(nW / 1024), 256, 0, stream>>>(Wv, wv16, (int)nW);
  cvt_f32_f16_k<<<(int)(nW / 1024), 256, 0, stream>>>(Wp, wp16, (int)nW);

  const int gemm_blocks = (M_ / 16) * (C_ / 64) / 4;   // 2048 blocks of 4 waves
  qkv_kernel<<<dim3(gemm_blocks, 3), 128, 0, stream>>>(
      x16, wq16, wk16, wv16, bq, bk, bv, rope, Qh, Kh, Vt);

  const int attn_blocks = (B_ * H_ * (T_ / 16)) / 4;   // 2048 blocks of 4 waves
  attn_kernel<<<attn_blocks, 128, 0, stream>>>(Qh, Kh, Vt, Yh);

  proj_kernel<<<gemm_blocks, 128, 0, stream>>>(Yh, wp16, bp, out);
}
